// VanillaRNN_69672959476113
// MI455X (gfx1250) — matrix operand order repaired and checked by reading, not compiled, not run on hardware
//
#include <hip/hip_runtime.h>
#include <hip/hip_bf16.h>

typedef __attribute__((ext_vector_type(16))) _Float16 v16h;
typedef __attribute__((ext_vector_type(8)))  _Float16 v8h;
typedef __attribute__((ext_vector_type(8)))  float    v8f;

#define HDIM 512
#define TSTEPS 512
#define BDIM 512
#define CDIM 10
#define HP 520        // padded LDS row stride (halves) -> spreads 64 banks
#define KT_REG 8      // kt 0..7  : B fragments held in VGPRs (128 regs/wave)
#define KT_LDS 8      // kt 8..15 : B fragments held in LDS   (256 KB/WGP)

// ---------------------------------------------------------------------------
// Pack Whh (f32, [K=512][N=512] row-major) into f16 WMMA B-fragment order.
// Fragment f = ntile*16 + ktile, 1 KB each (32 lanes x 16 halves):
//   lanes 0-15 : N = ntile*16 + lane,      K = ktile*32 + e        (e = 0..15)
//   lanes 16-31: N = ntile*16 + (lane-16), K = ktile*32 + 16 + e
// contiguous k order (not the ISA table order); A fragments are loaded in the same order.
// ---------------------------------------------------------------------------
__global__ void pack_whh_kernel(const float* __restrict__ Whh,
                                _Float16* __restrict__ pk) {
    int idx = blockIdx.x * blockDim.x + threadIdx.x;   // 0 .. 512*512-1
    if (idx >= HDIM * HDIM) return;
    int e    = idx & 15;
    int lane = (idx >> 4) & 31;
    int kt   = (idx >> 9) & 15;
    int nt   = idx >> 13;
    int k = kt * 32 + ((lane >> 4) << 4) + e;
    int n = nt * 16 + (lane & 15);
    pk[idx] = (_Float16)Whh[k * HDIM + n];
}

// ---------------------------------------------------------------------------
// One block = one independent 16-row batch tile, all 512 timesteps.
// 512 threads = 16 wave32s; wave w owns output column tiles {2w, 2w+1}.
// Whh lives half in VGPRs (kt 0..7) and half in LDS (kt 8..15): no global
// or scratch traffic inside the recurrence.
// ---------------------------------------------------------------------------
__global__ __launch_bounds__(512)
__attribute__((amdgpu_waves_per_eu(4, 4)))
void rnn_kernel(const float* __restrict__ x,      // [B][T]
                const float* __restrict__ Whx,    // [1][H]
                const _Float16* __restrict__ Bpk, // packed Whh fragments
                const float* __restrict__ bh,     // [H]
                const float* __restrict__ Wph,    // [H][C]
                const float* __restrict__ bp,     // [C]
                float* __restrict__ out)          // [B][C]
{
    __shared__ _Float16 ldsB[32 * KT_LDS * 512];  // 256 KB: frags kt=8..15
    __shared__ _Float16 hlds[16 * HP];            // ~17 KB: h tile, f16
    __shared__ __align__(32) float xs[16];        // x column for this tile

    const int tid   = threadIdx.x;
    const int lane  = tid & 31;
    const int wave  = tid >> 5;        // 0..15
    const int btile = blockIdx.x;      // 0..31
    const int hi16  = lane >> 4;       // 0 | 1
    const int row   = lane & 15;

    // ---- one-time LDS fill: fragments kt=8..15 for all 32 ntiles ----------
    {
        const uint4* src = (const uint4*)Bpk;     // 64 uint4 per 1 KB fragment
        uint4*       dst = (uint4*)ldsB;
        for (int c = tid; c < 32 * KT_LDS * 64; c += 512) {
            int frag   = c >> 6;                  // 0..255
            int within = c & 63;
            int nt  = frag >> 3;
            int kt8 = frag & 7;
            dst[c] = src[((nt * 16 + kt8 + KT_REG) << 6) + within];
        }
    }

    // ---- one-time register fill: fragments kt=0..7 for this wave's tiles --
    const int ntg0 = wave * 2, ntg1 = wave * 2 + 1;
    v16h Breg0[KT_REG], Breg1[KT_REG];
    #pragma unroll
    for (int kt = 0; kt < KT_REG; ++kt) {
        Breg0[kt] = *(const v16h*)&Bpk[(((ntg0 * 16 + kt) * 32) + lane) * 16];
        Breg1[kt] = *(const v16h*)&Bpk[(((ntg1 * 16 + kt) * 32) + lane) * 16];
    }

    // h0 = 0
    for (int i = tid; i < 16 * HP; i += 512) hlds[i] = (_Float16)0.f;

    // per-lane column constants
    const int n0 = ntg0 * 16 + row, n1 = ntg1 * 16 + row;
    const float whx0 = Whx[n0], whx1 = Whx[n1];
    const float bh0  = bh[n0],  bh1  = bh[n1];

    // preload x column for t = 0
    if (tid < 16) xs[tid] = x[(btile * 16 + tid) * TSTEPS + 0];
    __syncthreads();

    const _Float16* hrow = &hlds[row * HP];

    for (int t = 0; t < TSTEPS; ++t) {
        // accumulator init: xproj + bias (C/D layout: M = r + hi16*8)
        v8f xv = *(const v8f*)&xs[hi16 * 8];
        v8f acc0a = xv * whx0 + bh0;  v8f acc0b = {};
        v8f acc1a = xv * whx1 + bh1;  v8f acc1b = {};

        // ---- K loop, first half: B from registers ----
        #pragma unroll
        for (int kt = 0; kt < KT_REG; ++kt) {
            const int k0 = kt * 32;
            union { v16h v; v8h h[2]; } a;
            a.h[0] = *(const v8h*)&hrow[k0 + hi16 * 16];
            a.h[1] = *(const v8h*)&hrow[k0 + hi16 * 16 + 8];
            acc0a = __builtin_amdgcn_wmma_f32_16x16x32_f16(false, a.v, false, Breg0[kt],
                                                           (short)0, acc0a, false, false);
            acc1a = __builtin_amdgcn_wmma_f32_16x16x32_f16(false, a.v, false, Breg1[kt],
                                                           (short)0, acc1a, false, false);
        }
        // ---- K loop, second half: B from LDS ----
        #pragma unroll
        for (int kt8 = 0; kt8 < KT_LDS; ++kt8) {
            const int k0 = (KT_REG + kt8) * 32;
            union { v16h v; v8h h[2]; } a;
            a.h[0] = *(const v8h*)&hrow[k0 + hi16 * 16];
            a.h[1] = *(const v8h*)&hrow[k0 + hi16 * 16 + 8];
            const v16h b0 = *(const v16h*)&ldsB[(((ntg0 * KT_LDS + kt8) * 32) + lane) * 16];
            const v16h b1 = *(const v16h*)&ldsB[(((ntg1 * KT_LDS + kt8) * 32) + lane) * 16];
            acc0b = __builtin_amdgcn_wmma_f32_16x16x32_f16(false, a.v, false, b0,
                                                           (short)0, acc0b, false, false);
            acc1b = __builtin_amdgcn_wmma_f32_16x16x32_f16(false, a.v, false, b1,
                                                           (short)0, acc1b, false, false);
        }

        v8f acc0 = acc0a + acc0b;
        v8f acc1 = acc1a + acc1b;

        __syncthreads();   // all waves done READING hlds for step t

        // tanh -> f16 -> LDS (D layout: VGPR r -> M = r + hi16*8, N fixed)
        #pragma unroll
        for (int r = 0; r < 8; ++r) {
            const int m = r + hi16 * 8;
            float t0, t1;
            asm volatile("v_tanh_f32 %0, %1" : "=v"(t0) : "v"(acc0[r]));
            asm volatile("v_tanh_f32 %0, %1" : "=v"(t1) : "v"(acc1[r]));
            hlds[m * HP + n0] = (_Float16)t0;
            hlds[m * HP + n1] = (_Float16)t1;
        }
        // stage x column for t+1 (protected by the same barrier pair)
        if (tid < 16 && t + 1 < TSTEPS)
            xs[tid] = x[(btile * 16 + tid) * TSTEPS + (t + 1)];

        __syncthreads();   // writes visible before step t+1 reads
    }

    // ---- final projection: p = h_T @ Wph + bp  (16 x 10 per tile) ----
    if (tid < 16 * CDIM) {
        const int m = tid / CDIM, cc = tid % CDIM;
        float sum = bp[cc];
        for (int k = 0; k < HDIM; ++k)
            sum += (float)hlds[m * HP + k] * Wph[k * CDIM + cc];
        out[(btile * 16 + m) * CDIM + cc] = sum;
    }
}

// ---------------------------------------------------------------------------
extern "C" void kernel_launch(void* const* d_in, const int* in_sizes, int n_in,
                              void* d_out, int out_size, void* d_ws, size_t ws_size,
                              hipStream_t stream) {
    const float* x   = (const float*)d_in[0];   // [512,512]
    const float* Whx = (const float*)d_in[1];   // [1,512]
    const float* Whh = (const float*)d_in[2];   // [512,512]
    const float* Wph = (const float*)d_in[3];   // [512,10]
    const float* bh  = (const float*)d_in[4];   // [512]
    const float* bp  = (const float*)d_in[5];   // [10]
    float* out = (float*)d_out;                 // [512,10]

    _Float16* Bpk = (_Float16*)d_ws;            // 512 KB packed Whh (f16)

    pack_whh_kernel<<<(HDIM * HDIM + 255) / 256, 256, 0, stream>>>(Whh, Bpk);
    rnn_kernel<<<BDIM / 16, 512, 0, stream>>>(x, Whx, Bpk, bh, Wph, bp, out);
}
